// SumConv_54443005444407
// MI455X (gfx1250) — matrix operand order repaired and checked by reading, not compiled, not run on hardware
//
#include <hip/hip_runtime.h>

typedef __attribute__((ext_vector_type(16))) _Float16 v16h;
typedef __attribute__((ext_vector_type(8)))  _Float16 v8h;
typedef __attribute__((ext_vector_type(8)))  float    v8f;
typedef __attribute__((ext_vector_type(4)))  float    v4f;

#define FF 1024
#define CI 64
#define CO 64

static __device__ __forceinline__ v16h cat8(v8h a, v8h b) {
  return __builtin_shufflevector(a, b, 0,1,2,3,4,5,6,7,8,9,10,11,12,13,14,15);
}

// ---------------------------------------------------------------------------
// Kernel 1: P[cls][co][ci] = softmax_ci(logits[co,ci,kh,kw]) as f16.
// logits flat index: (co*64 + ci)*16 + cls, cls = kh*4+kw. 1024 threads.
// ---------------------------------------------------------------------------
__global__ void sumconv_softmax_prep(const float* __restrict__ logits,
                                     _Float16* __restrict__ P) {
  int t   = blockIdx.x * blockDim.x + threadIdx.x;   // 0..1023
  int co  = t >> 4;
  int cls = t & 15;
  const float* src = logits + co * (CI * 16) + cls;
  float mx = -3.4e38f;
  #pragma unroll 8
  for (int ci = 0; ci < CI; ++ci) mx = fmaxf(mx, src[ci * 16]);
  float s = 0.f;
  #pragma unroll 8
  for (int ci = 0; ci < CI; ++ci) s += __expf(src[ci * 16] - mx);
  float inv = 1.0f / s;
  _Float16* dst = P + (cls * CO + co) * CI;
  #pragma unroll 8
  for (int ci = 0; ci < CI; ++ci)
    dst[ci] = (_Float16)(__expf(src[ci * 16] - mx) * inv);
}

// ---------------------------------------------------------------------------
// Kernel 2: one pixel tile (16 columns) per wave. A = P tiles in VGPRs,
// E = exp(x-max) staged via LDS into WMMA B layout,
// D = P @ E with v_wmma_f32_16x16x32_f16, out = max + log(D).
// 256 blocks x 128 threads = 1024 waves = 16 classes x 64 tiles.
// ---------------------------------------------------------------------------
__global__ void __launch_bounds__(128)
sumconv_wmma(const float* __restrict__ ll, const _Float16* __restrict__ P,
             float* __restrict__ out) {
  __shared__ __align__(16) _Float16 ldsE[4][16][CI];

  int lane   = threadIdx.x & 31;
  int wv     = threadIdx.x >> 5;        // 0..3
  int wg     = blockIdx.x * 4 + wv;     // 0..1023
  int cls    = wg & 15;                 // kh*4 + kw
  int t      = wg >> 4;                 // tile within class, 0..63
  int kh     = cls >> 2, kw = cls & 3;
  int laneLo = lane & 15;
  int hi     = lane >> 4;               // half-wave select

  // ---- A operands: 4 M-tiles x 2 k-steps of P[cls], per ISA A-layout ----
  const _Float16* Pc = P + cls * (CO * CI);
  v16h Amat[4][2];
  #pragma unroll
  for (int mt = 0; mt < 4; ++mt) {
    const _Float16* row = Pc + (mt * 16 + laneLo) * CI;
    #pragma unroll
    for (int ks = 0; ks < 2; ++ks) {
      v8h lo = *(const v8h*)(row + ks * 32 + (hi ? 8  : 0));
      v8h hh = *(const v8h*)(row + ks * 32 + (hi ? 24 : 16));
      Amat[mt][ks] = cat8(lo, hh);
    }
  }

  // ---- this wave's 16 pixel columns ----
  int q  = t * 16 + laneLo;             // pixel id within class (column)
  int b  = q >> 6;
  int oh = (q >> 3) & 7;
  int ow = q & 7;
  int f  = (oh * 4 + kh) * 32 + (ow * 4 + kw);
  const float* xp = ll + ((size_t)(b * FF + f)) * CI + hi * 32;

  // this lane: 32 of the pixel's 64 ci values
  float xs[32];
  #pragma unroll
  for (int j = 0; j < 8; ++j) {
    v4f v = *(const v4f*)(xp + j * 4);
    xs[4*j] = v.x; xs[4*j+1] = v.y; xs[4*j+2] = v.z; xs[4*j+3] = v.w;
  }
  float mx = -3.4e38f;
  #pragma unroll
  for (int j = 0; j < 32; ++j) mx = fmaxf(mx, xs[j]);
  // combine the two half-wave partial maxes for this pixel
  mx = fmaxf(mx, __shfl_xor(mx, 16, 32));

  // E = exp(x - mx) as f16 -> LDS [pixel][ci]
  asm volatile("" ::: "memory");
  _Float16* ep = &ldsE[wv][laneLo][hi * 32];
  #pragma unroll
  for (int j = 0; j < 4; ++j) {
    v8h eh;
    #pragma unroll
    for (int k2 = 0; k2 < 8; ++k2)
      eh[k2] = (_Float16)__expf(xs[8*j + k2] - mx);
    *(v8h*)(ep + 8*j) = eh;
  }
  asm volatile("" ::: "memory");   // DS ops are in-order within a wave

  // B operands per ISA B-layout: lane = column(pixel), K packed 2/VGPR
  v16h Bmat[2];
  #pragma unroll
  for (int ks = 0; ks < 2; ++ks) {
    const _Float16* bp = &ldsE[wv][laneLo][ks * 32 + hi * 8];
    Bmat[ks] = cat8(*(const v8h*)bp, *(const v8h*)(bp + 16));
  }

  // D = P @ E ; out = mx + log(D)
  float* op = out + ((size_t)(b * FF + f)) * CO;
  #pragma unroll
  for (int mt = 0; mt < 4; ++mt) {
    v8f acc = {};
    acc = __builtin_amdgcn_wmma_f32_16x16x32_f16(
        false, Amat[mt][0], false, Bmat[0], (short)0, acc, false, false);
    acc = __builtin_amdgcn_wmma_f32_16x16x32_f16(
        false, Amat[mt][1], false, Bmat[1], (short)0, acc, false, false);
    int cobase = mt * 16 + hi * 8;      // D: VGPR v -> row M = v (+8 upper half)
    v4f r0, r1;
    r0.x = mx + __logf(acc[0]); r0.y = mx + __logf(acc[1]);
    r0.z = mx + __logf(acc[2]); r0.w = mx + __logf(acc[3]);
    r1.x = mx + __logf(acc[4]); r1.y = mx + __logf(acc[5]);
    r1.z = mx + __logf(acc[6]); r1.w = mx + __logf(acc[7]);
    *(v4f*)(op + cobase)     = r0;
    *(v4f*)(op + cobase + 4) = r1;
  }
}

extern "C" void kernel_launch(void* const* d_in, const int* in_sizes, int n_in,
                              void* d_out, int out_size, void* d_ws, size_t ws_size,
                              hipStream_t stream) {
  const float* ll     = (const float*)d_in[0];   // (16,1024,64,1) f32
  const float* logits = (const float*)d_in[1];   // (64,64,4,4,1) f32
  float*       out    = (float*)d_out;           // (16,1024,64,1) f32
  _Float16*    P      = (_Float16*)d_ws;         // 16*64*64 f16 = 128 KB

  sumconv_softmax_prep<<<4, 256, 0, stream>>>(logits, P);
  sumconv_wmma<<<256, 128, 0, stream>>>(ll, P, out);
}
